// DecoderLSTM_5282809774863
// MI455X (gfx1250) — matrix operand order repaired and checked by reading, not compiled, not run on hardware
//
#include <hip/hip_runtime.h>

// ---------------------------------------------------------------------------
// CDNA5 (gfx1250, wave32) batched LSTM decoder.
// gates[16,256] per wave-step via v_wmma_f32_16x16x32_f16.
// Bias folded into the weight matrix (K-row 130 with X[130]=1), so the first
// K-chunk accumulates onto the inline C=0 operand.
// Activations use native v_tanh_f32 when available.
// ---------------------------------------------------------------------------

typedef __attribute__((ext_vector_type(16))) _Float16 v16h;
typedef __attribute__((ext_vector_type(8)))  float    v8f;

#define HID   64
#define INSZ  66          // 2 pos + 64 context
#define BIASK 130         // K row holding the combined bias (X[130] == 1.0)
#define KPAD  160         // 2 + 64 + 64 + 1 bias + 29 zero pad (5 chunks of 32)
#define NG    256         // 4*H gate columns
#define PRED  12
#define KC    5           // K chunks
#define NT    16          // N tiles of 16
#define WAVES 8
#define SPW   16          // samples per wave
#define SPB   (WAVES * SPW)
#define WPACK_HALVES (KC * NT * 32 * 16)   // 40960 f16 = 80 KB

// ---------------------------------------------------------------------------
// Pack W = [W_ih^T ; W_hh^T ; bias ; 0] (160 x 256, f16) into per-lane WMMA
// B-fragment order: flat half index f = (((kc*16 + nt)*32 + lane)*16 + i):
//   n = nt*16 + (lane & 15),  k = kc*32 + (lane<16 ? 0 : 16) + i
// (contiguous k order, same as the A loads: lanes 0-15 hold K=0..15, lanes 16-31 K=16..31,
//  two K values packed per VGPR.)
// ---------------------------------------------------------------------------
__global__ void pack_weights_kernel(const float* __restrict__ W_ih,
                                    const float* __restrict__ W_hh,
                                    const float* __restrict__ b_ih,
                                    const float* __restrict__ b_hh,
                                    _Float16* __restrict__ wpack) {
    int f = blockIdx.x * blockDim.x + threadIdx.x;
    if (f >= WPACK_HALVES) return;
    int i    = f & 15;
    int lane = (f >> 4) & 31;
    int nt   = (f >> 9) & 15;
    int kc   = f >> 13;
    int n = nt * 16 + (lane & 15);
    int k = kc * 32 + ((lane < 16) ? 0 : 16) + i;
    float v = 0.0f;
    if (k < INSZ)            v = W_ih[n * INSZ + k];          // pos + context rows
    else if (k < INSZ + HID) v = W_hh[n * HID + (k - INSZ)];  // hidden rows
    else if (k == BIASK)     v = b_ih[n] + b_hh[n];           // folded bias row
    wpack[f] = (_Float16)v;
}

// ---- activations: prefer native v_tanh_f32 (gfx1250 transcendental) ----
__device__ __forceinline__ float fast_tanh(float x) {
#if __has_builtin(__builtin_amdgcn_tanhf)
    return __builtin_amdgcn_tanhf(x);
#else
    return 2.0f * __builtin_amdgcn_rcpf(1.0f + __expf(-2.0f * x)) - 1.0f;
#endif
}
__device__ __forceinline__ float fast_sigmoid(float x) {
#if __has_builtin(__builtin_amdgcn_tanhf)
    return 0.5f * __builtin_amdgcn_tanhf(0.5f * x) + 0.5f;
#else
    return __builtin_amdgcn_rcpf(1.0f + __expf(-x));
#endif
}

// ---------------------------------------------------------------------------
// Main kernel: 8 waves/block, 16 samples/wave. State (c in VGPRs, pos in LDS,
// h re-staged to LDS in f16) persists across the 12 recurrent steps.
// ---------------------------------------------------------------------------
__global__ __launch_bounds__(256) void
lstm_decoder_kernel(const float* __restrict__ context,
                    const float* __restrict__ start_pos,
                    const float* __restrict__ fc_w,
                    const float* __restrict__ fc_b,
                    const _Float16* __restrict__ wpack,
                    float* __restrict__ out,
                    int N) {
    __shared__ _Float16 sX[WAVES][SPW * KPAD];   // 40 KB: per-wave X staging
    __shared__ float    sPos[WAVES][SPW][2];

    const int tid  = threadIdx.x;
    const int wv   = tid >> 5;          // wave32
    const int lane = tid & 31;
    const int lhal = lane >> 4;         // lane half: 0 or 1
    const int l16  = lane & 15;
    const int base = blockIdx.x * SPB + wv * SPW;

    // ---- stage X = [pos | context | h=0 | 1.0 at BIASK | pad=0] as f16 ----
    for (int idx = lane; idx < SPW * HID; idx += 32) {
        int m = idx >> 6, kk = idx & 63;
        float v = (base + m < N) ? context[(size_t)(base + m) * HID + kk] : 0.0f;
        sX[wv][m * KPAD + 2 + kk] = (_Float16)v;
    }
    for (int idx = lane; idx < SPW * 2; idx += 32) {
        int m = idx >> 1, d = idx & 1;
        float p = (base + m < N) ? start_pos[(size_t)(base + m) * 2 + d] : 0.0f;
        sX[wv][m * KPAD + d] = (_Float16)p;
        sPos[wv][m][d] = p;
    }
    for (int idx = lane; idx < SPW * (KPAD - INSZ); idx += 32) {
        int m = idx / (KPAD - INSZ), kk = INSZ + idx % (KPAD - INSZ);
        sX[wv][m * KPAD + kk] = (_Float16)((kk == BIASK) ? 1.0f : 0.0f);
    }

    // ---- per-lane constants (kept in VGPRs across the whole kernel) ----
    float w0[4], w1[4];             // fc weights for hidden j = ntl*16 + l16
#pragma unroll
    for (int ntl = 0; ntl < 4; ++ntl) {
        int j = ntl * 16 + l16;
        w0[ntl] = fc_w[j];
        w1[ntl] = fc_w[HID + j];
    }
    const float fcb0 = fc_b[0], fcb1 = fc_b[1];

    float cst[4][8];                // cell state: (m = r + 8*lhal, j = ntl*16+l16)
#pragma unroll
    for (int ntl = 0; ntl < 4; ++ntl)
#pragma unroll
        for (int r = 0; r < 8; ++r) cst[ntl][r] = 0.0f;

    __syncthreads();

    const v16h* bp = (const v16h*)wpack;

#pragma unroll 1
    for (int step = 0; step < PRED; ++step) {
        // Compiler-level memory barrier: forbid hoisting the (loop-invariant)
        // B-fragment loads out of the step loop (else 640 VGPRs of hoisted
        // loads spill to scratch). B streams from L2 (80 KB, resident).
        asm volatile("" ::: "memory");

        // ---- A fragments for all 5 K-chunks (contiguous k = 16*lhal + s, same as B pack) ----
        const unsigned int* sxu = (const unsigned int*)&sX[wv][l16 * KPAD];
        union { v16h v; unsigned int u[8]; } afr[KC];
#pragma unroll
        for (int kc = 0; kc < KC; ++kc) {
            const int ka = kc * 32 + (lhal ? 16 : 0);   // VGPR 0..3
            const int kb = kc * 32 + (lhal ? 24 : 8);   // VGPR 4..7
#pragma unroll
            for (int j = 0; j < 4; ++j) afr[kc].u[j]     = sxu[(ka + 2 * j) >> 1];
#pragma unroll
            for (int j = 0; j < 4; ++j) afr[kc].u[4 + j] = sxu[(kb + 2 * j) >> 1];
        }

        // ---- gates = X(16x160) * W(160x256) via WMMA (C=0 on first chunk) ----
        v8f acc[NT];
#pragma unroll
        for (int kc = 0; kc < KC; ++kc) {
#pragma unroll
            for (int ntb = 0; ntb < NT / 4; ++ntb) {
                v16h bfr[4];
#pragma unroll
                for (int j = 0; j < 4; ++j)
                    bfr[j] = bp[(kc * NT + ntb * 4 + j) * 32 + lane];
#pragma unroll
                for (int j = 0; j < 4; ++j) {
                    const int nt = ntb * 4 + j;
                    v8f cin = (kc == 0) ? (v8f)(0.0f) : acc[nt];
                    acc[nt] = __builtin_amdgcn_wmma_f32_16x16x32_f16(
                        false, afr[kc].v, false, bfr[j], (short)0, cin,
                        false, false);
                }
            }
        }

        // ---- elementwise LSTM update; (lane, r, ntl) owns (m, j) ----
        float p0a[8], p1a[8];           // running fc partials per sample row
#pragma unroll
        for (int r = 0; r < 8; ++r) { p0a[r] = 0.0f; p1a[r] = 0.0f; }
#pragma unroll
        for (int ntl = 0; ntl < 4; ++ntl) {
#pragma unroll
            for (int r = 0; r < 8; ++r) {
                float is = fast_sigmoid(acc[ntl][r]);
                float fs = fast_sigmoid(acc[4 + ntl][r]);
                float gs = fast_tanh(acc[8 + ntl][r]);
                float os = fast_sigmoid(acc[12 + ntl][r]);
                float cn = fs * cst[ntl][r] + is * gs;
                cst[ntl][r] = cn;
                float hn = os * fast_tanh(cn);
                p0a[r] += hn * w0[ntl];
                p1a[r] += hn * w1[ntl];
                // restage h for next step's A matrix
                sX[wv][(r + lhal * 8) * KPAD + INSZ + ntl * 16 + l16] = (_Float16)hn;
            }
        }

        // ---- delta = h @ fc_w^T + fc_b ; pos += delta ; emit ----
#pragma unroll
        for (int r = 0; r < 8; ++r) {
            float p0 = p0a[r], p1 = p1a[r];
#pragma unroll
            for (int s = 1; s < 16; s <<= 1) {   // butterfly within 16-lane half
                p0 += __shfl_xor(p0, s, 32);
                p1 += __shfl_xor(p1, s, 32);
            }
            int m = r + lhal * 8;
            if (l16 == 0 && base + m < N) {
                float nx = sPos[wv][m][0] + p0 + fcb0;
                float ny = sPos[wv][m][1] + p1 + fcb1;
                sPos[wv][m][0] = nx;
                sPos[wv][m][1] = ny;
                sX[wv][m * KPAD + 0] = (_Float16)nx;
                sX[wv][m * KPAD + 1] = (_Float16)ny;
                size_t o = ((size_t)(base + m) * PRED + step) * 2;
                out[o]     = nx;
                out[o + 1] = ny;
            }
        }
        __syncthreads();   // h/pos restage visible before next step's A loads
    }
}

// ---------------------------------------------------------------------------
extern "C" void kernel_launch(void* const* d_in, const int* in_sizes, int n_in,
                              void* d_out, int out_size, void* d_ws, size_t ws_size,
                              hipStream_t stream) {
    const float* context   = (const float*)d_in[0];
    const float* start_pos = (const float*)d_in[1];
    const float* W_ih      = (const float*)d_in[2];
    const float* W_hh      = (const float*)d_in[3];
    const float* b_ih      = (const float*)d_in[4];
    const float* b_hh      = (const float*)d_in[5];
    const float* fc_w      = (const float*)d_in[6];
    const float* fc_b      = (const float*)d_in[7];
    float* out = (float*)d_out;

    const int N = in_sizes[0] / HID;

    _Float16* wpack = (_Float16*)d_ws;

    hipLaunchKernelGGL(pack_weights_kernel,
                       dim3((WPACK_HALVES + 255) / 256), dim3(256), 0, stream,
                       W_ih, W_hh, b_ih, b_hh, wpack);

    const int blocks = (N + SPB - 1) / SPB;
    hipLaunchKernelGGL(lstm_decoder_kernel,
                       dim3(blocks), dim3(256), 0, stream,
                       context, start_pos, fc_w, fc_b, wpack, out, N);
}
